// VQVAE_9672266350815
// MI455X (gfx1250) — matrix operand order repaired and checked by reading, not compiled, not run on hardware
//
#include <hip/hip_runtime.h>
#include <hip/hip_bf16.h>

typedef __attribute__((ext_vector_type(16))) __bf16 v16bf;
typedef __attribute__((ext_vector_type(8)))  __bf16 v8bf;
typedef __attribute__((ext_vector_type(8)))  float  v8f;
typedef __attribute__((ext_vector_type(2)))  float  v2f;

#if __has_builtin(__builtin_amdgcn_global_load_async_to_lds_b32)
#define USE_ASYNC_LDS 1
#else
#define USE_ASYNC_LDS 0
#endif

typedef __attribute__((address_space(1))) int  gint_t;
typedef __attribute__((address_space(1))) const float gcfloat_t;
typedef __attribute__((address_space(3))) int  lint_t;
typedef __attribute__((address_space(3))) float lfloat_t;

// ---------------------------------------------------------------------------
// Implicit-GEMM convolution (normal + transposed) with WMMA bf16.
//   M = output channels (32 per block, LDS-staged weights shared by 4 waves)
//   N = flattened (batch, oh, ow)  (16 per wave, 64 per block)
//   K = Ci*KH*KW in steps of 32
// Weight panel is staged global->LDS with async-to-LDS ops (ASYNCcnt), then
// converted f32->bf16 into the WMMA-layout panel.
// KH/KW/TRANSPOSED/STRIDE are compile-time so all index math is mul/shift.
// Ho/Wo are powers of two for every layer -> mask/shift pixel decomposition.
// ---------------------------------------------------------------------------
struct ConvP {
    const float* x; const float* w; const float* bias; const float* res; float* y;
    int N, Ci, H, W, Co, Ho, Wo, pad;
    int lwo, lho;           // log2(Wo), log2(Ho)
    int relu_in, relu_out;
};

#define STAGE_K 256         // K-chunk per async staging round (32x256 f32 = 32 KB)

template<int KH, int KW, int TR, int S>
__global__ __launch_bounds__(128) void conv_wmma(ConvP p) {
    extern __shared__ __align__(16) __bf16 sA[];   // [32][Kpad] bf16 weight panel
    __shared__ __align__(16) float sStage[32 * STAGE_K];

    constexpr int KHW = KH * KW;
    const int Ktot = p.Ci * KHW;
    const int Kpad = (Ktot + 31) & ~31;
    const int co0  = blockIdx.y * 32;

    // ---- stage weight panel: global -> LDS (async) -> bf16 panel ----------
    for (int kc = 0; kc < Kpad; kc += STAGE_K) {
        // 1) async copy the valid f32 elements of this K-chunk into sStage
        for (int e = threadIdx.x; e < 32 * STAGE_K; e += 128) {
            const int r  = e >> 8;              // row 0..31 (STAGE_K == 256)
            const int k  = kc + (e & (STAGE_K - 1));
            const int mm = co0 + r;
            if (k < Ktot && mm < p.Co) {
                const int ci = k / KHW;         // compile-time divisor
                const int rr = k - ci * KHW;
                const size_t widx = TR
                    ? ((size_t)ci * p.Co + mm) * KHW + rr   // torch convT: w[in][out][kh][kw]
                    : ((size_t)mm * p.Ci + ci) * KHW + rr;  // conv:        w[out][in][kh][kw]
#if USE_ASYNC_LDS
                // two-step casts: addrspacecast first, then pointee/const cast
                gcfloat_t* gsrcf = (gcfloat_t*)(p.w + widx);
                lfloat_t*  ldstf = (lfloat_t*)&sStage[e];
                __builtin_amdgcn_global_load_async_to_lds_b32(
                    (gint_t*)gsrcf, (lint_t*)ldstf, 0, 0);
#else
                sStage[e] = p.w[widx];
#endif
            }
        }
#if USE_ASYNC_LDS
        asm volatile("s_wait_asynccnt 0x0" ::: "memory");
#endif
        __syncthreads();
        // 2) convert chunk to bf16 WMMA panel (zero-fill K/Co padding)
        for (int e = threadIdx.x; e < 32 * STAGE_K; e += 128) {
            const int r = e >> 8;
            const int k = kc + (e & (STAGE_K - 1));
            if (k < Kpad) {
                const bool valid = (k < Ktot) && ((co0 + r) < p.Co);
                sA[r * Kpad + k] = static_cast<__bf16>(valid ? sStage[e] : 0.0f);
            }
        }
        __syncthreads();
    }

    const int lane  = threadIdx.x & 31;
    const int wv    = threadIdx.x >> 5;
    const int ntile = blockIdx.x * 4 + wv;          // wave-uniform
    const int Ntot  = p.N << (p.lwo + p.lho);
    if (ntile * 16 >= Ntot) return;                 // uniform per wave (after barriers)

    const int m    = lane & 15;
    const int half = lane >> 4;

    // B column -> (b, oh, ow) via shifts/masks
    const int  n      = ntile * 16 + (lane & 15);
    const bool nvalid = n < Ntot;
    const int  nc = nvalid ? n : 0;
    const int  ow = nc & ((1 << p.lwo) - 1);
    const int  oh = (nc >> p.lwo) & ((1 << p.lho) - 1);
    const int  bb = nc >> (p.lwo + p.lho);
    const float* xb = p.x + (size_t)bb * p.Ci * p.H * p.W;

    const __bf16* arow0 = sA + (size_t)m        * Kpad;
    const __bf16* arow1 = sA + (size_t)(m + 16) * Kpad;

    v8f acc0 = {};
    v8f acc1 = {};
    for (int k0 = 0; k0 < Ktot; k0 += 32) {
        // ---- B: activation tile 32x16 bf16 (per-lane gather) ----------
        // lane col = lane&15; VGPR v holds rows k = (v<4?0:16) + half*8 + (v&3)*2 (+e)
        v16bf Bf;
        #pragma unroll
        for (int v = 0; v < 8; ++v) {
            #pragma unroll
            for (int e = 0; e < 2; ++e) {
                const int k = k0 + (v < 4 ? 0 : 16) + half * 8 + (v & 3) * 2 + e;
                float val = 0.0f;
                if (k < Ktot && nvalid) {
                    const int ci = k / KHW;
                    const int rr = k - ci * KHW;
                    const int kh = rr / KW;
                    const int kw = rr - kh * KW;
                    int ih, iw; bool ok;
                    if (!TR) {
                        ih = oh * S - p.pad + kh;
                        iw = ow * S - p.pad + kw;
                        ok = (unsigned)ih < (unsigned)p.H && (unsigned)iw < (unsigned)p.W;
                    } else {
                        const int th = oh + p.pad - kh;
                        const int tw = ow + p.pad - kw;
                        ok = th >= 0 && tw >= 0 &&
                             (th % S) == 0 && (tw % S) == 0;   // S compile-time
                        ih = th / S; iw = tw / S;
                        ok = ok && ih < p.H && iw < p.W;
                    }
                    if (ok) {
                        float xv = xb[((size_t)ci * p.H + ih) * p.W + iw];
                        if (p.relu_in) xv = fmaxf(xv, 0.0f);
                        val = xv;
                    }
                }
                Bf[v * 2 + e] = static_cast<__bf16>(val);
            }
        }

        // ---- A: two 16x32 fragments from LDS (2x 16B loads each) ------
        // element v*2+e <-> k = (v<4?0:16) + half*8 + (v&3)*2 + e
        //   => elements 0..7  = k0 + half*8 .. +8      (contiguous)
        //      elements 8..15 = k0 + 16 + half*8 .. +8 (contiguous)
        const v8bf a0lo = *(const v8bf*)(arow0 + k0 + half * 8);
        const v8bf a0hi = *(const v8bf*)(arow0 + k0 + 16 + half * 8);
        const v8bf a1lo = *(const v8bf*)(arow1 + k0 + half * 8);
        const v8bf a1hi = *(const v8bf*)(arow1 + k0 + 16 + half * 8);
        v16bf A0, A1;
        #pragma unroll
        for (int i2 = 0; i2 < 8; ++i2) {
            A0[i2] = a0lo[i2]; A0[8 + i2] = a0hi[i2];
            A1[i2] = a1lo[i2]; A1[8 + i2] = a1hi[i2];
        }
        acc0 = __builtin_amdgcn_wmma_f32_16x16x32_bf16(false, A0, false, Bf,
                                                       (short)0, acc0, false, false);
        acc1 = __builtin_amdgcn_wmma_f32_16x16x32_bf16(false, A1, false, Bf,
                                                       (short)0, acc1, false, false);
    }

    // ---- epilogue: lane holds col n for rows r + 8*half of each tile ---
    if (!nvalid) return;
    #pragma unroll
    for (int tile = 0; tile < 2; ++tile) {
        #pragma unroll
        for (int r = 0; r < 8; ++r) {
            const int mm = co0 + tile * 16 + r + 8 * half;
            if (mm < p.Co) {
                float val = (tile == 0) ? acc0[r] : acc1[r];
                if (p.bias) val += p.bias[mm];
                const size_t oidx = (((size_t)bb * p.Co + mm) << (p.lho + p.lwo))
                                  + ((size_t)oh << p.lwo) + ow;
                if (p.res) val += p.res[oidx];
                if (p.relu_out) val = fmaxf(val, 0.0f);
                p.y[oidx] = val;
            }
        }
    }
}

// ---------------------------------------------------------------------------
// Codebook squared norms: cnorm[j] = sum_d cb[j][d]^2    (512 x 64)
// ---------------------------------------------------------------------------
__global__ void cb_norms(const float* __restrict__ cb, float* __restrict__ cnorm) {
    const int j = blockIdx.x * blockDim.x + threadIdx.x;
    if (j < 512) {
        float s = 0.0f;
        #pragma unroll 8
        for (int d = 0; d < 64; ++d) { const float v = cb[j * 64 + d]; s += v * v; }
        cnorm[j] = s;
    }
}

__global__ void zero1(float* p) { p[0] = 0.0f; }

// ---------------------------------------------------------------------------
// VQ argmin via f32 WMMA (16x16x4): scores = cnorm[j] - 2 * <flat_i, cb_j>
// ---------------------------------------------------------------------------
__global__ __launch_bounds__(128) void vq_argmin(const float* __restrict__ ze,
                                                 const float* __restrict__ cb,
                                                 const float* __restrict__ cnorm,
                                                 int* __restrict__ idx) {
    const int D = 64;
    const int lane  = threadIdx.x & 31;
    const int wv    = threadIdx.x >> 5;
    const int mtile = blockIdx.x * 4 + wv;      // 4096 tiles over 65536 pixels
    const int half  = lane >> 4;
    const int m     = lane & 15;
    const int ncol  = lane & 15;

    // pixel i -> (b, h, w);  flat[i][d] = ze[((b*64 + d)*64 + h)*64 + w]
    const int i = mtile * 16 + m;
    const int w = i & 63, h = (i >> 6) & 63, b = i >> 12;

    // Preload A operand values for all 16 K-steps.
    // A layout (32-bit 16x4): lane row = lane&15; VGPR v -> K = half*2 + v.
    float areg[32];
    #pragma unroll
    for (int s = 0; s < 16; ++s) {
        const int k = s * 4 + half * 2;
        const size_t base = (((size_t)b * D + k) * 64 + h) * 64 + w;
        areg[2 * s]     = ze[base];
        areg[2 * s + 1] = ze[base + 4096];      // d+1  -> +H*W
    }

    float best[8]; int bj[8];
    #pragma unroll
    for (int r = 0; r < 8; ++r) { best[r] = 3.4e38f; bj[r] = 0; }

    for (int j0 = 0; j0 < 512; j0 += 16) {
        v8f acc = {};
        const float* cbn = cb + (size_t)(j0 + ncol) * D;
        #pragma unroll
        for (int s = 0; s < 16; ++s) {
            v2f a, bb;
            a[0] = areg[2 * s]; a[1] = areg[2 * s + 1];
            const int k = s * 4 + half * 2;     // B (4x16): VGPR v -> row half*2 + v
            bb[0] = cbn[k]; bb[1] = cbn[k + 1];
            acc = __builtin_amdgcn_wmma_f32_16x16x4_f32(false, a, false, bb,
                                                        (short)0, acc, false, false);
        }
        const float cn = cnorm[j0 + ncol];
        const int   jj = j0 + ncol;
        #pragma unroll
        for (int r = 0; r < 8; ++r) {
            const float sc = cn - 2.0f * acc[r];
            if (sc < best[r] || (sc == best[r] && jj < bj[r])) { best[r] = sc; bj[r] = jj; }
        }
    }
    // reduce over the 16 columns (stays within each 16-lane half)
    #pragma unroll
    for (int off = 1; off < 16; off <<= 1) {
        #pragma unroll
        for (int r = 0; r < 8; ++r) {
            const float ob = __shfl_xor(best[r], off, 32);
            const int   oj = __shfl_xor(bj[r],   off, 32);
            if (ob < best[r] || (ob == best[r] && oj < bj[r])) { best[r] = ob; bj[r] = oj; }
        }
    }
    if (ncol == 0) {
        #pragma unroll
        for (int r = 0; r < 8; ++r) {
            idx[mtile * 16 + r + 8 * half] = bj[r];   // C rows: r + 8*half
        }
    }
}

// ---------------------------------------------------------------------------
// Gather z_q (NCHW) + accumulate sum((z_q - z_e)^2) via wave reduction.
// Forward loss value = (1 + beta) * mean((z_q - z_e)^2).
// ---------------------------------------------------------------------------
__global__ void vq_gather_loss(const float* __restrict__ cb, const int* __restrict__ idx,
                               const float* __restrict__ ze, float* __restrict__ zq,
                               float* __restrict__ acc, int total) {
    const int t = blockIdx.x * blockDim.x + threadIdx.x;
    float d = 0.0f;
    if (t < total) {
        // t = ((b*64 + dch)*64 + h)*64 + w  over [16,64,64,64]
        const int w  = t & 63;
        const int h  = (t >> 6)  & 63;
        const int dc = (t >> 12) & 63;
        const int b  = t >> 18;
        const int i  = b * 4096 + h * 64 + w;
        const int j  = idx[i];
        const float q = cb[j * 64 + dc];
        const float e = ze[t];
        zq[t] = q;
        const float df = q - e;
        d = df * df;
    }
    for (int off = 16; off; off >>= 1) d += __shfl_down(d, off, 32);
    if ((threadIdx.x & 31) == 0) atomicAdd(acc, d);
}

__global__ void finalize_loss(float* out, const float* acc) {
    out[0] = 1.25f * acc[0] * (1.0f / 4194304.0f);   // (1+beta) * mean over 16*64*64*64
}

// ---------------------------------------------------------------------------
// Host orchestration
// ---------------------------------------------------------------------------
template<int KH, int KW, int TR, int S>
static void conv_call(const float* x, const float* w, const float* b, const float* res,
                      float* y, int N, int Ci, int H, int W, int Co, int Ho, int Wo,
                      int pad, int rin, int rout, hipStream_t stream) {
    ConvP p;
    p.x = x; p.w = w; p.bias = b; p.res = res; p.y = y;
    p.N = N; p.Ci = Ci; p.H = H; p.W = W; p.Co = Co; p.Ho = Ho; p.Wo = Wo;
    p.pad = pad; p.lwo = __builtin_ctz(Wo); p.lho = __builtin_ctz(Ho);
    p.relu_in = rin; p.relu_out = rout;
    const int Ntot = N * Ho * Wo;
    const int nt   = (Ntot + 15) / 16;
    dim3 grid((nt + 3) / 4, (Co + 31) / 32);
    const int Ktot = Ci * KH * KW;
    const int Kpad = (Ktot + 31) & ~31;
    const size_t shmem = (size_t)32 * Kpad * sizeof(__bf16);
    hipLaunchKernelGGL((conv_wmma<KH, KW, TR, S>), grid, dim3(128), shmem, stream, p);
}

extern "C" void kernel_launch(void* const* d_in, const int* in_sizes, int n_in,
                              void* d_out, int out_size, void* d_ws, size_t ws_size,
                              hipStream_t stream) {
    const float* x      = (const float*)d_in[0];
    const float* enc_w1 = (const float*)d_in[1];
    const float* enc_b1 = (const float*)d_in[2];
    const float* enc_w2 = (const float*)d_in[3];
    const float* enc_b2 = (const float*)d_in[4];
    const float* enc_w3 = (const float*)d_in[5];
    const float* enc_b3 = (const float*)d_in[6];
    const float* enc_rw1= (const float*)d_in[7];
    const float* enc_rw2= (const float*)d_in[8];
    const float* pre_w  = (const float*)d_in[9];
    const float* pre_b  = (const float*)d_in[10];
    const float* cbk    = (const float*)d_in[11];
    const float* dec_wt1= (const float*)d_in[12];
    const float* dec_bt1= (const float*)d_in[13];
    const float* dec_rw1= (const float*)d_in[14];
    const float* dec_rw2= (const float*)d_in[15];
    const float* dec_wt2= (const float*)d_in[16];
    const float* dec_bt2= (const float*)d_in[17];
    const float* dec_wt3= (const float*)d_in[18];
    const float* dec_bt3= (const float*)d_in[19];

    // ---- workspace layout (floats), heavy reuse -----------------------
    float* ws  = (float*)d_ws;
    float* H0  = ws;                         // 8,388,608 f (32 MB): res ping / z3
    float* H1  = ws + 8388608;               // 8,388,608 f (32 MB): res pong
    float* Z1  = ws;                         // 16,777,216 f (64 MB = H0+H1): enc conv1 out
    float* B1  = ws + 16777216;              // 16,777,216 f (64 MB): z2 / h / z_e / z_q / d2
    float* z2   = B1;
    float* henc = B1;
    float* ze   = B1;
    float* zq   = B1 + 4194304;
    float* hdec = B1;
    float* d2   = B1;
    float* tail  = ws + 33554432;
    int*   idx   = (int*)tail;               // 65536 ints
    float* cnorm = tail + 65536;             // 512 f
    float* lacc  = cnorm + 512;              // 1 f

    // ---- encoder ------------------------------------------------------
    conv_call<4,4,0,2>(x,  enc_w1, enc_b1, nullptr, Z1, 16,   3, 256, 256,  64, 128, 128, 1, 0, 1, stream);
    conv_call<4,4,0,2>(Z1, enc_w2, enc_b2, nullptr, z2, 16,  64, 128, 128, 128,  64,  64, 1, 0, 1, stream);
    conv_call<3,3,0,1>(z2, enc_w3, enc_b3, nullptr, H0, 16, 128,  64,  64, 128,  64,  64, 1, 0, 0, stream);
    {   // 2 shared resblocks, ping-pong H0/H1
        float* P = H0; float* Q = H1;
        for (int i = 0; i < 2; ++i) {
            conv_call<3,3,0,1>(P,    enc_rw1, nullptr, nullptr, henc, 16, 128, 64, 64,  32, 64, 64, 1, 1, 0, stream);
            conv_call<1,1,0,1>(henc, enc_rw2, nullptr, P,       Q,    16,  32, 64, 64, 128, 64, 64, 0, 1, 0, stream);
            float* t = P; P = Q; Q = t;
        }
    }
    // res_stack final relu fused into pre-vq conv input
    conv_call<1,1,0,1>(H0, pre_w, pre_b, nullptr, ze, 16, 128, 64, 64, 64, 64, 64, 0, 1, 0, stream);

    // ---- quantizer ----------------------------------------------------
    hipLaunchKernelGGL(cb_norms, dim3(2), dim3(256), 0, stream, cbk, cnorm);
    hipLaunchKernelGGL(zero1, dim3(1), dim3(1), 0, stream, lacc);
    hipLaunchKernelGGL(vq_argmin, dim3(1024), dim3(128), 0, stream, ze, cbk, cnorm, idx);
    hipLaunchKernelGGL(vq_gather_loss, dim3(4194304 / 256), dim3(256), 0, stream,
                       cbk, idx, ze, zq, lacc, 4194304);

    // ---- decoder ------------------------------------------------------
    conv_call<3,3,1,1>(zq, dec_wt1, dec_bt1, nullptr, H0, 16, 64, 64, 64, 128, 64, 64, 1, 0, 0, stream);
    {   // 32 shared resblocks (bug-faithful), ping-pong H0/H1
        float* P = H0; float* Q = H1;
        for (int i = 0; i < 32; ++i) {
            conv_call<3,3,0,1>(P,    dec_rw1, nullptr, nullptr, hdec, 16, 128, 64, 64,   2, 64, 64, 1, 1, 0, stream);
            conv_call<1,1,0,1>(hdec, dec_rw2, nullptr, P,       Q,    16,   2, 64, 64, 128, 64, 64, 0, 1, 0, stream);
            float* t = P; P = Q; Q = t;
        }
    }
    // res relu fused into conv_t2 input; conv_t2 output relu'd
    conv_call<4,4,1,2>(H0, dec_wt2, dec_bt2, nullptr, d2, 16, 128, 64, 64, 64, 128, 128, 1, 1, 1, stream);
    float* xhat = (float*)d_out;
    conv_call<4,4,1,2>(d2, dec_wt3, dec_bt3, nullptr, xhat, 16, 64, 128, 128, 3, 256, 256, 1, 1, 0, stream);

    hipLaunchKernelGGL(finalize_loss, dim3(1), dim3(1), 0, stream,
                       xhat + (out_size - 1), lacc);
}
